// MAB_68719477328
// MI455X (gfx1250) — matrix operand order repaired and checked by reading, not compiled, not run on hardware
//
#include <hip/hip_runtime.h>

typedef _Float16 half_t;
typedef __attribute__((ext_vector_type(16))) _Float16 v16h;
typedef __attribute__((ext_vector_type(8)))  _Float16 v8h;
typedef __attribute__((ext_vector_type(8)))  float    v8f;
typedef __attribute__((ext_vector_type(4)))  float    v4f;
typedef int v4i_t __attribute__((__vector_size__(16)));

#define WMMA_F16(a, b, c) \
  __builtin_amdgcn_wmma_f32_16x16x32_f16(false, (a), false, (b), (short)0, (c), false, false)

// ---------------- async global->LDS copy (CDNA5 ASYNCcnt path), guarded ----
// clang signature: void(V4i addrspace(1)*, V4i addrspace(3)*, Imm int off, Imm int cpol)
#if __has_builtin(__builtin_amdgcn_global_load_async_to_lds_b128)
#define HAVE_ASYNC_LDS 1
// generic->AS1 / AS3 via integer round-trip: global generic == AS1 numerically;
// generic LDS pointers carry the byte offset in the low 32 bits.
#define ASYNC_CP(ldst, gsrc)                                                        \
  __builtin_amdgcn_global_load_async_to_lds_b128(                                   \
      (__attribute__((address_space(1))) v4i_t*)(unsigned long long)(gsrc),         \
      (__attribute__((address_space(3))) v4i_t*)(unsigned)(unsigned long long)(ldst), \
      0, 0)
#else
#define HAVE_ASYNC_LDS 0
#define ASYNC_CP(ldst, gsrc) (*(v8h*)(ldst) = *(const v8h*)(gsrc))
#endif

static __device__ __forceinline__ void async_wait_all() {
#if HAVE_ASYNC_LDS
#if __has_builtin(__builtin_amdgcn_s_wait_asynccnt)
  __builtin_amdgcn_s_wait_asynccnt(0);
#else
  asm volatile("s_wait_asynccnt 0x0" ::: "memory");
#endif
#endif
}

// ---------------- fragment builders -----------------------------------------
static __device__ __forceinline__ v16h cat8(v8h lo, v8h hi) {
  return __builtin_shufflevector(lo, hi, 0,1,2,3,4,5,6,7,8,9,10,11,12,13,14,15);
}

// A fragment (16x32 f16): lane L holds row m = L%16, K = (L/16)*8 + {0..7, 16..23}
static __device__ __forceinline__ v16h load_a_frag(const half_t* base, int ld, int lane) {
  const half_t* p = base + (size_t)(lane & 15) * ld + ((lane >> 4) << 3);
  v8h lo = *(const v8h*)(p);
  v8h hi = *(const v8h*)(p + 16);
  return cat8(lo, hi);
}

// Same A fragment but converting from f32 source in-register
static __device__ __forceinline__ v16h load_a_frag_f32(const float* base, int ld, int lane) {
  const float* p = base + (size_t)(lane & 15) * ld + ((lane >> 4) << 3);
  v4f a0 = *(const v4f*)(p);
  v4f a1 = *(const v4f*)(p + 4);
  v4f b0 = *(const v4f*)(p + 16);
  v4f b1 = *(const v4f*)(p + 20);
  v16h r;
#pragma unroll
  for (int j = 0; j < 4; ++j) {
    r[j]      = (half_t)a0[j];
    r[4 + j]  = (half_t)a1[j];
    r[8 + j]  = (half_t)b0[j];
    r[12 + j] = (half_t)b1[j];
  }
  return r;
}

// B fragment (32x16 f16): lane L holds column n = L%16, K = (L/16)*8 + {0..7, 16..23}.
// base points at element [n0][k0] of a row-major matrix whose ROWS are B's columns.
static __device__ __forceinline__ v16h load_b_frag(const half_t* base, int ld, int lane) {
  const half_t* p = base + (size_t)(lane & 15) * ld + ((lane >> 4) << 3);
  v8h lo = *(const v8h*)(p);
  v8h hi = *(const v8h*)(p + 16);
  return cat8(lo, hi);
}

// ---------------- 16-lane XOR-butterfly max (VALU permlane path) -------------
static __device__ __forceinline__ float red_max16(float x) {
#if __has_builtin(__builtin_amdgcn_permlane16)
  // v_permlane16_b32: src1 = selects for row lanes 0..7, src2 = lanes 8..15
  unsigned u;
  u = __float_as_uint(x);
  x = fmaxf(x, __uint_as_float(__builtin_amdgcn_permlane16(u, u, 0x67452301u, 0xEFCDAB89u, false, false))); // ^1
  u = __float_as_uint(x);
  x = fmaxf(x, __uint_as_float(__builtin_amdgcn_permlane16(u, u, 0x54761032u, 0xDCFE98BAu, false, false))); // ^2
  u = __float_as_uint(x);
  x = fmaxf(x, __uint_as_float(__builtin_amdgcn_permlane16(u, u, 0x32107654u, 0xBA98FEDCu, false, false))); // ^4
  u = __float_as_uint(x);
  x = fmaxf(x, __uint_as_float(__builtin_amdgcn_permlane16(u, u, 0xFEDCBA98u, 0x76543210u, false, false))); // ^8
#else
#pragma unroll
  for (int off = 1; off < 16; off <<= 1) x = fmaxf(x, __shfl_xor(x, off, 16));
#endif
  return x;
}

// ---------------- converts ---------------------------------------------------
__global__ __launch_bounds__(256) void cvt_f32_f16(const float* __restrict__ s,
                                                   half_t* __restrict__ d, int n8) {
  int i = blockIdx.x * 256 + threadIdx.x;
  if (i >= n8) return;
  const v4f* sp = (const v4f*)s + (size_t)i * 2;
  v4f a = sp[0], b = sp[1];
  v8h o;
#pragma unroll
  for (int j = 0; j < 4; ++j) { o[j] = (half_t)a[j]; o[4 + j] = (half_t)b[j]; }
  *((v8h*)d + i) = o;
}

// ---------------- projection GEMM -------------------------------------------
// C[8192][512] = A[8192][512] * W^T + bias.  W row-major [512][512] f16.
// MODE 0: f16 row-major. MODE 1: per-head transposed Vt[((b*8+h)*64+hd)*2048+q].
// MODE 2: f32 row-major.
template <int MODE, bool AF32>
__global__ __launch_bounds__(256) void proj_gemm(const void* __restrict__ Ap,
                                                 const half_t* __restrict__ W,
                                                 const float* __restrict__ bias,
                                                 void* __restrict__ outp) {
  const int lane = threadIdx.x & 31;
  const int wave = threadIdx.x >> 5;
  const int m0 = blockIdx.x * 128 + wave * 16;
  const int n0 = blockIdx.y * 64;

  v8f zero = {};
  v8f acc[4];
#pragma unroll
  for (int t = 0; t < 4; ++t) acc[t] = zero;

  const half_t* wrow = W + (size_t)n0 * 512;
  for (int k0 = 0; k0 < 512; k0 += 32) {
    v16h af;
    if (AF32) af = load_a_frag_f32((const float*)Ap + (size_t)m0 * 512 + k0, 512, lane);
    else      af = load_a_frag((const half_t*)Ap + (size_t)m0 * 512 + k0, 512, lane);
    if (k0 + 32 < 512) __builtin_prefetch((const half_t*)wrow + k0 + 32, 0, 3);
#pragma unroll
    for (int nt = 0; nt < 4; ++nt) {
      v16h bf = load_b_frag(wrow + (size_t)nt * 16 * 512 + k0, 512, lane);
      acc[nt] = WMMA_F16(af, bf, acc[nt]);
    }
  }

  const int nlane = lane & 15;
  const int hs = lane >> 4;  // C layout: VGPR i <-> (m = m0 + i + 8*hs, n = n0 + nt*16 + nlane)
  if (MODE == 0 || MODE == 2) {
#pragma unroll
    for (int nt = 0; nt < 4; ++nt) {
      const int n = n0 + nt * 16 + nlane;
      const float bv = bias[n];
#pragma unroll
      for (int i = 0; i < 8; ++i) {
        size_t idx = (size_t)(m0 + 8 * hs + i) * 512 + n;
        float v = acc[nt][i] + bv;
        if (MODE == 0) ((half_t*)outp)[idx] = (half_t)v;
        else           ((float*)outp)[idx] = v;
      }
    }
  } else {  // MODE 1: transposed per-head store, one contiguous v8h per lane
    const int b = m0 >> 11;
    const int q = (m0 & 2047) + 8 * hs;
#pragma unroll
    for (int nt = 0; nt < 4; ++nt) {
      const int n = n0 + nt * 16 + nlane;
      const float bv = bias[n];
      v8h pk;
#pragma unroll
      for (int i = 0; i < 8; ++i) pk[i] = (half_t)(acc[nt][i] + bv);
      half_t* dst = (half_t*)outp + ((size_t)(b * 8 + (n >> 6)) * 64 + (n & 63)) * 2048 + q;
      *(v8h*)dst = pk;
    }
  }
}

// ---------------- flash attention -------------------------------------------
// O = Q + softmax(Q K^T / 8) V per (b,h).  Qh,Kh row-major f16 [8192][512];
// Vt per-head transposed f16 [32][64][2048]; Oh f16 row-major [8192][512].
// 256 threads = 8 waves; block = 128 queries; K/V tiles (64 keys) staged in LDS
// via async copies, double-buffered, shared by all 8 waves.
__global__ __launch_bounds__(256) void attn_kernel(const half_t* __restrict__ Qh,
                                                   const half_t* __restrict__ Kh,
                                                   const half_t* __restrict__ Vt,
                                                   half_t* __restrict__ Oh) {
  __shared__ __align__(16) half_t ktile[2][64][72];  // [buf][key][hd],   rows padded
  __shared__ __align__(16) half_t vtile[2][64][72];  // [buf][hd][key],   rows padded
  __shared__ __align__(16) half_t pbuf[8][16][72];   // per-wave P transpose tile

  const int lane = threadIdx.x & 31;
  const int wave = threadIdx.x >> 5;
  const int nlane = lane & 15;
  const int hs = lane >> 4;
  const int bh = blockIdx.y;
  const int b = bh >> 3, h = bh & 7;
  const int q0 = blockIdx.x * 128 + wave * 16;

  const half_t* kgbase = Kh + (size_t)(b * 2048) * 512 + h * 64;
  const half_t* vbh = Vt + (size_t)bh * 64 * 2048;

  const half_t* qbase = Qh + (size_t)(b * 2048 + q0) * 512 + h * 64;
  v16h qf0 = load_a_frag(qbase, 512, lane);       // head-dim 0..31
  v16h qf1 = load_a_frag(qbase + 32, 512, lane);  // head-dim 32..63

  // B-fragment with a single column (n==0) of ones: P x onesb accumulates row sums.
  v16h onesb;
#pragma unroll
  for (int j = 0; j < 16; ++j) onesb[j] = (nlane == 0) ? (half_t)1.0f : (half_t)0.0f;

  v8f zero = {};
  v8f oacc[4];
#pragma unroll
  for (int t = 0; t < 4; ++t) oacc[t] = zero;
  v8f sacc = zero;  // running softmax denominators (column 0)
  v8f mrow;
#pragma unroll
  for (int i = 0; i < 8; ++i) mrow[i] = -3.0e38f;

  half_t* pb = &pbuf[wave][0][0];

  // issue async copies of K/V tile `kt` into buffer `buf` (whole block cooperates)
  auto issue_tile = [&](int kt, int buf) {
    const half_t* kg = kgbase + (size_t)kt * 512;
    const half_t* vg = vbh + kt;
#pragma unroll
    for (int c = 0; c < 2; ++c) {
      int idx = (int)threadIdx.x + 256 * c;  // 0..511 chunk id
      int row = idx >> 3;
      int ch = (idx & 7) * 8;
      ASYNC_CP(&ktile[buf][row][ch], kg + (size_t)row * 512 + ch);
      ASYNC_CP(&vtile[buf][row][ch], vg + (size_t)row * 2048 + ch);
    }
  };

  issue_tile(0, 0);
  int buf = 0;
  for (int kt = 0; kt < 2048; kt += 64, buf ^= 1) {
    async_wait_all();
    __syncthreads();                         // tile `buf` visible to all waves;
    if (kt + 64 < 2048) issue_tile(kt + 64, buf ^ 1);  // prefetch next tile

    // ---- S = Q K^T * (1/sqrt(64)) for 64 keys (from LDS K tile)
    const half_t* kb = &ktile[buf][0][0];
    v8f s[4];
#pragma unroll
    for (int nt = 0; nt < 4; ++nt) {
      v16h kf0 = load_b_frag(kb + nt * 16 * 72, 72, lane);
      v16h kf1 = load_b_frag(kb + nt * 16 * 72 + 32, 72, lane);
      v8f z = zero;
      z = WMMA_F16(qf0, kf0, z);
      z = WMMA_F16(qf1, kf1, z);
#pragma unroll
      for (int i = 0; i < 8; ++i) s[nt][i] = z[i] * 0.125f;
    }

    // ---- online softmax: row max via VALU permlane butterfly
    v8f corr;
#pragma unroll
    for (int i = 0; i < 8; ++i) {
      float t = fmaxf(fmaxf(s[0][i], s[1][i]), fmaxf(s[2][i], s[3][i]));
      t = red_max16(t);
      float nm = fmaxf(mrow[i], t);
      corr[i] = __expf(mrow[i] - nm);
      mrow[i] = nm;
    }
#pragma unroll
    for (int nt = 0; nt < 4; ++nt)
#pragma unroll
      for (int i = 0; i < 8; ++i) oacc[nt][i] *= corr[i];
#pragma unroll
    for (int i = 0; i < 8; ++i) sacc[i] *= corr[i];
#pragma unroll
    for (int nt = 0; nt < 4; ++nt)
#pragma unroll
      for (int i = 0; i < 8; ++i) s[nt][i] = __expf(s[nt][i] - mrow[i]);

    // ---- C-layout -> A-layout transpose of probabilities via per-wave LDS tile
#pragma unroll
    for (int nt = 0; nt < 4; ++nt)
#pragma unroll
      for (int i = 0; i < 8; ++i)
        pbuf[wave][8 * hs + i][nt * 16 + nlane] = (half_t)s[nt][i];
    v16h pf0 = load_a_frag(pb, 72, lane);       // keys 0..31 of tile
    v16h pf1 = load_a_frag(pb + 32, 72, lane);  // keys 32..63 of tile

    // ---- O += P V (from LDS V tile); row sums via ones-column WMMA
#pragma unroll
    for (int nt = 0; nt < 4; ++nt) {
      v16h vf0 = load_b_frag(&vtile[buf][nt * 16][0], 72, lane);
      v16h vf1 = load_b_frag(&vtile[buf][nt * 16][32], 72, lane);
      oacc[nt] = WMMA_F16(pf0, vf0, oacc[nt]);
      oacc[nt] = WMMA_F16(pf1, vf1, oacc[nt]);
    }
    sacc = WMMA_F16(pf0, onesb, sacc);
    sacc = WMMA_F16(pf1, onesb, sacc);

    __syncthreads();  // all waves done reading `buf` before it is refilled
  }

  // ---- epilogue: broadcast denominators, normalize, residual Q, store
  v8f lrow;
#pragma unroll
  for (int i = 0; i < 8; ++i) lrow[i] = __shfl(sacc[i], lane & 16, 32);
#pragma unroll
  for (int nt = 0; nt < 4; ++nt) {
    const int hd = nt * 16 + nlane;
#pragma unroll
    for (int i = 0; i < 8; ++i) {
      const int q = q0 + 8 * hs + i;
      size_t row = (size_t)(b * 2048 + q) * 512 + h * 64 + hd;
      float o = oacc[nt][i] / lrow[i] + (float)Qh[row];
      Oh[row] = (half_t)o;
    }
  }
}

// ---------------- launch -----------------------------------------------------
extern "C" void kernel_launch(void* const* d_in, const int* in_sizes, int n_in,
                              void* d_out, int out_size, void* d_ws, size_t ws_size,
                              hipStream_t stream) {
  const float* X  = (const float*)d_in[0];
  const float* Y  = (const float*)d_in[1];
  const float* Wq = (const float*)d_in[2];
  const float* bq = (const float*)d_in[3];
  const float* Wk = (const float*)d_in[4];
  const float* bk = (const float*)d_in[5];
  const float* Wv = (const float*)d_in[6];
  const float* bv = (const float*)d_in[7];
  const float* Wo = (const float*)d_in[8];
  const float* bo = (const float*)d_in[9];

  const size_t MAT = 8192 * 512;  // 4,194,304
  const size_t WEL = 512 * 512;   // 262,144

  half_t* Qh  = (half_t*)d_ws;
  half_t* Kh  = Qh + MAT;
  half_t* Vt  = Kh + MAT;
  half_t* Oh  = Vt + MAT;
  half_t* Wqh = Oh + MAT;
  half_t* Wkh = Wqh + WEL;
  half_t* Wvh = Wkh + WEL;
  half_t* Woh = Wvh + WEL;

  {
    int n8 = (int)(WEL / 8);
    int blk = (n8 + 255) / 256;
    cvt_f32_f16<<<blk, 256, 0, stream>>>(Wq, Wqh, n8);
    cvt_f32_f16<<<blk, 256, 0, stream>>>(Wk, Wkh, n8);
    cvt_f32_f16<<<blk, 256, 0, stream>>>(Wv, Wvh, n8);
    cvt_f32_f16<<<blk, 256, 0, stream>>>(Wo, Woh, n8);
  }

  dim3 ggrid(64, 8);  // 8192/128 x 512/64
  proj_gemm<0, true><<<ggrid, 256, 0, stream>>>((const void*)X, Wqh, bq, (void*)Qh);
  proj_gemm<0, true><<<ggrid, 256, 0, stream>>>((const void*)Y, Wkh, bk, (void*)Kh);
  proj_gemm<1, true><<<ggrid, 256, 0, stream>>>((const void*)Y, Wvh, bv, (void*)Vt);

  attn_kernel<<<dim3(16, 32), 256, 0, stream>>>(Qh, Kh, Vt, Oh);

  proj_gemm<2, false><<<ggrid, 256, 0, stream>>>((const void*)Oh, Woh, bo, d_out);
}
